// GCN_89928025244112
// MI455X (gfx1250) — matrix operand order repaired and checked by reading, not compiled, not run on hardware
//
#include <hip/hip_runtime.h>
#include <hip/hip_bf16.h>

// ---------------------------------------------------------------------------
// GCN forward for MI455X (gfx1250, wave32, WMMA).
// GEMMs: v_wmma_f32_16x16x32_bf16, bf16 operands pre-packed in memory so the
// inner loop is pure b128 loads + wmma (no conversion VALU). Each wave owns a
// 16x64 output tile (A fragment reused across 4 WMMAs).
// Edge propagation: coalesced float4 gathers + native f32 global atomics.
// Workspace: ~260 MB.
// ---------------------------------------------------------------------------

typedef __attribute__((ext_vector_type(16))) __bf16 v16bf;
typedef __attribute__((ext_vector_type(8)))  __bf16 v8bf;
typedef __attribute__((ext_vector_type(8)))  float  v8f;

#define D_H   256
#define D_OUT 128
#define EPSL  1e-5f

using idx_t = int;  // harness passes integer inputs as int32

__device__ __forceinline__ __bf16 f2bf(float f) {
  union { float f; unsigned u; } x; x.f = f;
  unsigned r = (x.u + 0x7FFFu + ((x.u >> 16) & 1u)) >> 16;  // RNE truncate
  unsigned short s = (unsigned short)r;
  __bf16 o; __builtin_memcpy(&o, &s, 2);
  return o;
}

__device__ __forceinline__ void atomAdd(float* p, float v) {
  unsafeAtomicAdd(p, v);   // native global_atomic_add_f32
}

// ---------------------------------------------------------------------------
// WMMA GEMM: one wave computes a 16x64 tile of C = A[M,K] @ W[K,Nc].
// A: bf16 row-major.  Per ISA 7.12.2 the A fragment per lane is two
// contiguous 8-element K-chunks: [8*half, +8) and [16+8*half, +8)  -> 2xb128.
// W: pre-swizzled bf16 so each lane's 16 B elements are contiguous:
//   Wsw[((nt*ksteps + ks)*32 + lane)*16 + t] -> K=ks*32+8*half+(t&7)+16*(t>>3),
//   col = nt*16 + lane%16                                        -> 2xb128.
// D: acc[r] = D[half*8 + r][lane%16].
// ---------------------------------------------------------------------------
template<int RELU, int BIAS, int OUT_BF16>
__global__ __launch_bounds__(256)
void gemm_wmma(const __bf16* __restrict__ A, const __bf16* __restrict__ Wsw,
               const float* __restrict__ bias, float* __restrict__ C,
               __bf16* __restrict__ Cbf, int M, int K, int Nc, int totalWaves) {
  int wid = blockIdx.x * 8 + (threadIdx.x >> 5);
  if (wid >= totalWaves) return;                 // wave-uniform: EXEC all-ones
  int lane = threadIdx.x & 31;
  int ngrp = Nc >> 6;                            // 64-column groups
  int mt = wid / ngrp, ng = wid - mt * ngrp;
  int half = lane >> 4, l16 = lane & 15;
  int row = mt * 16 + l16;
  int ksteps = K >> 5;
  const __bf16* arow = A + (size_t)row * K + 8 * half;
  v8f acc[4] = {v8f{}, v8f{}, v8f{}, v8f{}};
  for (int ks = 0; ks < ksteps; ++ks) {
    v8bf a0 = *(const v8bf*)(arow + ks * 32);
    v8bf a1 = *(const v8bf*)(arow + ks * 32 + 16);
    v16bf a;
#pragma unroll
    for (int j = 0; j < 8; ++j) { a[j] = a0[j]; a[j + 8] = a1[j]; }
#pragma unroll
    for (int t = 0; t < 4; ++t) {
      v16bf b = *(const v16bf*)(Wsw +
          ((size_t)(ng * 4 + t) * ksteps + ks) * 512 + lane * 16);
      acc[t] = __builtin_amdgcn_wmma_f32_16x16x32_bf16(
          false, a, false, b, (short)0, acc[t], false, false);
    }
  }
  int orow = mt * 16 + half * 8;
#pragma unroll
  for (int t = 0; t < 4; ++t) {
    int col = (ng * 4 + t) * 16 + l16;
    float bv = BIAS ? bias[col] : 0.0f;
#pragma unroll
    for (int r = 0; r < 8; ++r) {
      float v = acc[t][r] + bv;
      if (RELU) v = fmaxf(v, 0.0f);
      if (OUT_BF16) Cbf[(size_t)(orow + r) * Nc + col] = f2bf(v);
      else          C  [(size_t)(orow + r) * Nc + col] = v;
    }
  }
}

// Swizzle+convert weights into the per-lane-contiguous B fragment layout.
__global__ void wswz_k(const float* __restrict__ W, __bf16* __restrict__ Wsw,
                       int K, int Nc, long long total) {
  long long gid = (long long)blockIdx.x * blockDim.x + threadIdx.x;
  if (gid >= total) return;
  int t    = (int)(gid & 15);
  int lane = (int)((gid >> 4) & 31);
  int f    = (int)(gid >> 9);
  int ksteps = K >> 5;
  int nt = f / ksteps, ks = f - nt * ksteps;
  int j = t >> 1, lo = t & 1, half = lane >> 4;
  int kk  = ks * 32 + 8 * half + (t & 7) + 16 * (t >> 3);
  int col = nt * 16 + (lane & 15);
  Wsw[gid] = f2bf(W[(size_t)kk * Nc + col]);
}

__global__ void cvt_bf_k(const float* __restrict__ in, __bf16* __restrict__ out,
                         long long n) {
  long long i = (long long)blockIdx.x * blockDim.x + threadIdx.x;
  if (i < n) out[i] = f2bf(in[i]);
}

// ---------------------------------------------------------------------------
// Graph / normalization kernels
// ---------------------------------------------------------------------------
__global__ void fill_k(float* __restrict__ p, float val, long long n) {
  long long i = (long long)blockIdx.x * blockDim.x + threadIdx.x;
  if (i < n) p[i] = val;
}

__global__ void deg_acc_k(const idx_t* __restrict__ col, const float* __restrict__ w,
                          float* __restrict__ deg, int E) {
  int e = blockIdx.x * blockDim.x + threadIdx.x;
  if (e < E) atomAdd(&deg[col[e]], w[e]);
}

__global__ void rsqrt_k(float* __restrict__ p, int n) {
  int i = blockIdx.x * blockDim.x + threadIdx.x;
  if (i < n) { float d = p[i]; p[i] = d > 0.f ? rsqrtf(d) : 0.f; }
}

__global__ void norm_k(const idx_t* __restrict__ row, const idx_t* __restrict__ col,
                       const float* __restrict__ w, const float* __restrict__ dis,
                       float* __restrict__ nrm, int E) {
  int e = blockIdx.x * blockDim.x + threadIdx.x;
  if (e < E) nrm[e] = dis[row[e]] * w[e] * dis[col[e]];
}

// agg = dis[n]^2 * xw   (self-loop term of the normalized adjacency)
__global__ void selfloop_k(const float* __restrict__ dis, const float* __restrict__ xw,
                           float* __restrict__ agg, long long total, int cshift) {
  long long i = (long long)blockIdx.x * blockDim.x + threadIdx.x;
  if (i >= total) return;
  int node = (int)(i >> cshift);
  float d = dis[node];
  agg[i] = d * d * xw[i];
}

// Edge scatter-add: thread = (edge, 4-channel group); channels fastest.
__global__ __launch_bounds__(256)
void scatter_k(const idx_t* __restrict__ row, const idx_t* __restrict__ col,
               const float* __restrict__ nrm, const float* __restrict__ xw,
               float* __restrict__ agg, int E, int cshift2, int C) {
  long long gid = (long long)blockIdx.x * blockDim.x + threadIdx.x;
  int e  = (int)(gid >> cshift2);
  int cg = (int)(gid & ((1 << cshift2) - 1));
  if (e >= E) return;
  int r = row[e], c = col[e];
  float nv = nrm[e];
  float4 v = *(const float4*)(xw + (size_t)r * C + cg * 4);
  float* dst = agg + (size_t)c * C + cg * 4;
  atomAdd(dst + 0, nv * v.x);
  atomAdd(dst + 1, nv * v.y);
  atomAdd(dst + 2, nv * v.z);
  atomAdd(dst + 3, nv * v.w);
}

// Wave-per-row LayerNorm over 256 channels (8 per lane, shfl_xor reduce).
__global__ __launch_bounds__(256)
void layernorm_k(const float* __restrict__ in, const float* __restrict__ g,
                 const float* __restrict__ b, float* __restrict__ out, int n) {
  int widx = blockIdx.x * 8 + (threadIdx.x >> 5);
  if (widx >= n) return;
  int lane = threadIdx.x & 31;
  const float* r = in + (size_t)widx * D_H;
  float v[8]; float s = 0.f;
#pragma unroll
  for (int j = 0; j < 8; ++j) { v[j] = r[lane + 32 * j]; s += v[j]; }
#pragma unroll
  for (int o = 16; o > 0; o >>= 1) s += __shfl_xor(s, o, 32);
  float mu = s * (1.0f / D_H);
  float ss = 0.f;
#pragma unroll
  for (int j = 0; j < 8; ++j) { float d = v[j] - mu; ss += d * d; }
#pragma unroll
  for (int o = 16; o > 0; o >>= 1) ss += __shfl_xor(ss, o, 32);
  float rs = rsqrtf(ss * (1.0f / D_H) + EPSL);
  float* w = out + (size_t)widx * D_H;
#pragma unroll
  for (int j = 0; j < 8; ++j) {
    int c = lane + 32 * j;
    w[c] = (v[j] - mu) * rs * g[c] + b[c];
  }
}

// BatchNorm stats: block accumulates 128 rows locally, one atomic per channel.
__global__ __launch_bounds__(256)
void bn_stats_k(const float* __restrict__ x, float* __restrict__ sums,
                float* __restrict__ sumsq, int n) {
  int c = threadIdx.x;
  int r0 = blockIdx.x * 128;
  int rend = min(r0 + 128, n);
  float s = 0.f, q = 0.f;
  for (int r = r0; r < rend; ++r) {
    float v = x[(size_t)r * D_H + c];
    s += v; q += v * v;
  }
  atomAdd(&sums[c], s);
  atomAdd(&sumsq[c], q);
}

__global__ void bn_finalize_k(const float* __restrict__ sums, const float* __restrict__ sumsq,
                              const float* __restrict__ g, const float* __restrict__ b,
                              float* __restrict__ scale, float* __restrict__ shift, int n) {
  int c = threadIdx.x;
  float inv = 1.0f / (float)n;
  float mean = sums[c] * inv;
  float var  = sumsq[c] * inv - mean * mean;   // biased var, matches reference
  float sc = g[c] * rsqrtf(var + EPSL);
  scale[c] = sc;
  shift[c] = b[c] - mean * sc;
}

// cur = relu(agg*scale+shift) + cur (+ nf on layer 0); also emit bf16 copy
// (next layer's GEMM A operand).
__global__ void bn_apply_k(const float* __restrict__ agg, const float* __restrict__ scale,
                           const float* __restrict__ shift, const float* __restrict__ nf,
                           float* __restrict__ cur, __bf16* __restrict__ curbf,
                           long long total, int addNF) {
  long long i = (long long)blockIdx.x * blockDim.x + threadIdx.x;
  if (i >= total) return;
  int c = (int)(i & (D_H - 1));
  float v = fmaxf(agg[i] * scale[c] + shift[c], 0.f) + cur[i];
  if (addNF) v += nf[i];
  cur[i] = v;
  curbf[i] = f2bf(v);
}

// node_emb = agg + bias -> d_out ; accumulate column sums for mean pool
__global__ void final_k(const float* __restrict__ agg, const float* __restrict__ bias,
                        float* __restrict__ out, float* __restrict__ pool, long long total) {
  long long i = (long long)blockIdx.x * blockDim.x + threadIdx.x;
  if (i >= total) return;
  int c = (int)(i & (D_OUT - 1));
  float v = agg[i] + bias[c];
  out[i] = v;
  atomAdd(&pool[c], v);
}

__global__ void pool_div_k(const float* __restrict__ pool, float* __restrict__ out, int n) {
  int c = threadIdx.x;   // 128 threads
  out[(size_t)n * D_OUT + c] = pool[c] / (float)n;
}

// ---------------------------------------------------------------------------
extern "C" void kernel_launch(void* const* d_in, const int* in_sizes, int n_in,
                              void* d_out, int out_size, void* d_ws, size_t ws_size,
                              hipStream_t stream) {
  const float* x       = (const float*)d_in[0];
  const idx_t* ei      = (const idx_t*)d_in[1];
  const float* ew      = (const float*)d_in[2];
  const float* W_nt1   = (const float*)d_in[3];
  const float* b_nt1   = (const float*)d_in[4];
  const float* W_nt2   = (const float*)d_in[5];
  const float* b_nt2   = (const float*)d_in[6];
  const float* ln_g    = (const float*)d_in[7];
  const float* ln_b    = (const float*)d_in[8];
  const float* Wc      = (const float*)d_in[9];
  // d_in[10] == bc: per-channel conv bias cancels exactly inside BatchNorm
  const float* bn_g    = (const float*)d_in[11];
  const float* bn_b    = (const float*)d_in[12];
  const float* Wc_last = (const float*)d_in[13];
  const float* bc_last = (const float*)d_in[14];

  const int N = in_sizes[0] / D_H;   // 50000
  const int E = in_sizes[2];         // 800000
  const idx_t* row = ei;
  const idx_t* col = ei + E;

  // ---- workspace carve-out (f32 region then bf16 region) --------------
  float* wsf   = (float*)d_ws;
  float* dis   = wsf;  wsf += N;
  float* nrm   = wsf;  wsf += E;
  float* nf    = wsf;  wsf += (size_t)N * D_H;
  float* cur   = wsf;  wsf += (size_t)N * D_H;
  float* xw    = wsf;  wsf += (size_t)N * D_H;
  float* agg   = wsf;  wsf += (size_t)N * D_H;
  float* sums  = wsf;  wsf += D_H;   // sums & sumsq contiguous
  float* sumsq = wsf;  wsf += D_H;
  float* scale = wsf;  wsf += D_H;
  float* shift = wsf;  wsf += D_H;
  float* pool  = wsf;  wsf += D_OUT;
  __bf16* wsb  = (__bf16*)wsf;
  __bf16* xbf    = wsb;  wsb += (size_t)N * D_H;
  __bf16* curbf  = wsb;  wsb += (size_t)N * D_H;   // also holds h1 bf16
  __bf16* W1sw   = wsb;  wsb += D_H * D_H;
  __bf16* W2sw   = wsb;  wsb += D_H * D_H;
  __bf16* WcSw   = wsb;  wsb += 3 * D_H * D_H;
  __bf16* WlSw   = wsb;  wsb += D_H * D_OUT;

  auto cdiv = [](long long a, long long b) { return (int)((a + b - 1) / b); };

  // ---- operand preparation (bf16 convert + B-fragment swizzle) --------
  cvt_bf_k<<<cdiv((long long)N * D_H, 256), 256, 0, stream>>>(x, xbf, (long long)N * D_H);
  const long long wsz = (long long)D_H * D_H;
  wswz_k<<<cdiv(wsz, 256), 256, 0, stream>>>(W_nt1, W1sw, D_H, D_H, wsz);
  wswz_k<<<cdiv(wsz, 256), 256, 0, stream>>>(W_nt2, W2sw, D_H, D_H, wsz);
  for (int L = 0; L < 3; ++L)
    wswz_k<<<cdiv(wsz, 256), 256, 0, stream>>>(Wc + (size_t)L * wsz,
                                               WcSw + (size_t)L * wsz, D_H, D_H, wsz);
  const long long wlz = (long long)D_H * D_OUT;
  wswz_k<<<cdiv(wlz, 256), 256, 0, stream>>>(Wc_last, WlSw, D_H, D_OUT, wlz);

  // ---- symmetric normalization: D^{-1/2}(A+I)D^{-1/2} -----------------
  fill_k<<<cdiv(N, 256), 256, 0, stream>>>(dis, 1.0f, N);
  deg_acc_k<<<cdiv(E, 256), 256, 0, stream>>>(col, ew, dis, E);
  rsqrt_k<<<cdiv(N, 256), 256, 0, stream>>>(dis, N);
  norm_k<<<cdiv(E, 256), 256, 0, stream>>>(row, col, ew, dis, nrm, E);

  // ---- node_transform: relu(x@W1+b1)@W2+b2 -> LayerNorm ---------------
  const int wv256 = (N / 16) * (D_H / 64);
  gemm_wmma<1, 1, 1><<<cdiv(wv256, 8), 256, 0, stream>>>(
      xbf, W1sw, b_nt1, nullptr, curbf, N, D_H, D_H, wv256);       // h1 (bf16)
  gemm_wmma<0, 1, 0><<<cdiv(wv256, 8), 256, 0, stream>>>(
      curbf, W2sw, b_nt2, agg, nullptr, N, D_H, D_H, wv256);       // h2 (f32)
  layernorm_k<<<cdiv(N, 8), 256, 0, stream>>>(agg, ln_g, ln_b, nf, N);

  // ---- cur = x --------------------------------------------------------
  hipMemcpyAsync(cur, x, (size_t)N * D_H * sizeof(float),
                 hipMemcpyDeviceToDevice, stream);

  // ---- 3 hidden GCNConv + BN + ReLU + residual ------------------------
  for (int L = 0; L < 3; ++L) {
    const __bf16* Ain = (L == 0) ? xbf : curbf;   // cur==x on layer 0
    gemm_wmma<0, 0, 0><<<cdiv(wv256, 8), 256, 0, stream>>>(
        Ain, WcSw + (size_t)L * wsz, nullptr, xw, nullptr, N, D_H, D_H, wv256);
    selfloop_k<<<cdiv((long long)N * D_H, 256), 256, 0, stream>>>(
        dis, xw, agg, (long long)N * D_H, 8);
    scatter_k<<<cdiv((long long)E * (D_H / 4), 256), 256, 0, stream>>>(
        row, col, nrm, xw, agg, E, 6, D_H);
    fill_k<<<2, 256, 0, stream>>>(sums, 0.f, 2 * D_H);
    bn_stats_k<<<cdiv(N, 128), 256, 0, stream>>>(agg, sums, sumsq, N);
    bn_finalize_k<<<1, D_H, 0, stream>>>(sums, sumsq, bn_g + L * D_H, bn_b + L * D_H,
                                         scale, shift, N);
    bn_apply_k<<<cdiv((long long)N * D_H, 256), 256, 0, stream>>>(
        agg, scale, shift, nf, cur, curbf, (long long)N * D_H, L == 0 ? 1 : 0);
  }

  // ---- final GCNConv 256 -> 128, + bias, + mean pool ------------------
  const int wv128 = (N / 16) * (D_OUT / 64);
  gemm_wmma<0, 0, 0><<<cdiv(wv128, 8), 256, 0, stream>>>(
      curbf, WlSw, nullptr, xw, nullptr, N, D_H, D_OUT, wv128);
  selfloop_k<<<cdiv((long long)N * D_OUT, 256), 256, 0, stream>>>(
      dis, xw, agg, (long long)N * D_OUT, 7);
  scatter_k<<<cdiv((long long)E * (D_OUT / 4), 256), 256, 0, stream>>>(
      row, col, nrm, xw, agg, E, 5, D_OUT);
  fill_k<<<1, D_OUT, 0, stream>>>(pool, 0.f, D_OUT);
  final_k<<<cdiv((long long)N * D_OUT, 256), 256, 0, stream>>>(
      agg, bc_last, (float*)d_out, pool, (long long)N * D_OUT);
  pool_div_k<<<1, D_OUT, 0, stream>>>(pool, (float*)d_out, N);
}
